// DeepseekV2DecoderLayer_44616120271592
// MI455X (gfx1250) — matrix operand order repaired and checked by reading, not compiled, not run on hardware
//
// DeepSeek-V2 decoder layer for MI455X (gfx1250) — bf16 WMMA everywhere.
#include <hip/hip_runtime.h>
#include <stdint.h>

typedef __attribute__((ext_vector_type(16))) __bf16 v16bf;
typedef __attribute__((ext_vector_type(8)))  float  v8f;
typedef unsigned int u32;

// ---------------------------------------------------------------- helpers
__device__ __forceinline__ __bf16 f2bf(float f) {
  u32 u = __builtin_bit_cast(u32, f);
  u32 r = (u + 0x7FFFu + ((u >> 16) & 1u)) >> 16;
  unsigned short s = (unsigned short)r;
  return __builtin_bit_cast(__bf16, s);
}

union Frag { uint4 q[2]; v16bf v; };

__device__ __forceinline__ v8f wmma_bf16(v16bf a, v16bf b, v8f c) {
  return __builtin_amdgcn_wmma_f32_16x16x32_bf16(false, a, false, b, (short)0, c, false, false);
}

// ============================================================== transpose
// in: f32 [R][C] (batched), out: bf16 [C][R]  (B^T layout for WMMA B frags)
__global__ __launch_bounds__(256) void transpose_f32_bf16(
    const float* __restrict__ in, __bf16* __restrict__ out, int R, int C) {
  __shared__ float tile[32][33];
  const long bo = (long)blockIdx.z * (long)R * (long)C;
  in += bo; out += bo;
  const int r0 = blockIdx.y * 32, c0 = blockIdx.x * 32;
  const int tx = threadIdx.x & 31, ty = threadIdx.x >> 5; // 32 x 8
  #pragma unroll
  for (int i = ty; i < 32; i += 8) {
    int r = r0 + i, c = c0 + tx;
    tile[i][tx] = (r < R && c < C) ? in[(long)r * C + c] : 0.0f;
  }
  __syncthreads();
  #pragma unroll
  for (int i = ty; i < 32; i += 8) {
    int c = c0 + i, r = r0 + tx;
    if (c < C && r < R) out[(long)c * R + r] = f2bf(tile[tx][i]);
  }
}

// ============================================================== rmsnorm
__global__ __launch_bounds__(256) void rmsnorm_bf16_kernel(
    const float* __restrict__ x, const float* __restrict__ wgt,
    __bf16* __restrict__ out, int width) {
  const int row = blockIdx.x;
  const float* xr = x + (long)row * width;
  float ss = 0.0f;
  for (int i = threadIdx.x; i < width; i += 256) { float v = xr[i]; ss += v * v; }
  #pragma unroll
  for (int o = 16; o > 0; o >>= 1) ss += __shfl_xor(ss, o, 32);
  __shared__ float red[8];
  __shared__ float invs;
  const int w = threadIdx.x >> 5, lane = threadIdx.x & 31;
  if (lane == 0) red[w] = ss;
  __syncthreads();
  if (w == 0) {
    float t = (lane < 8) ? red[lane] : 0.0f;
    #pragma unroll
    for (int o = 4; o > 0; o >>= 1) t += __shfl_xor(t, o, 32);
    if (lane == 0) invs = rsqrtf(t / (float)width + 1e-6f);
  }
  __syncthreads();
  const float inv = invs;
  __bf16* orow = out + (long)row * width;
  for (int i = threadIdx.x; i < width; i += 256) orow[i] = f2bf(xr[i] * inv * wgt[i]);
}

// ======================================== kv: rmsnorm(k_lat)+rope(k_pe)
// kv f32 (S,576) -> kf bf16 (S,576), k_latT bf16 (512,S)
__global__ __launch_bounds__(256) void kv_norm_rope_kernel(
    const float* __restrict__ kv, const float* __restrict__ lnw,
    const float* __restrict__ cosb, const float* __restrict__ sinb,
    __bf16* __restrict__ kf, __bf16* __restrict__ klT, int S) {
  const int s = blockIdx.x;
  const float* r = kv + (long)s * 576;
  float ss = 0.0f;
  for (int i = threadIdx.x; i < 512; i += 256) { float v = r[i]; ss += v * v; }
  #pragma unroll
  for (int o = 16; o > 0; o >>= 1) ss += __shfl_xor(ss, o, 32);
  __shared__ float red[8];
  __shared__ float invs;
  const int w = threadIdx.x >> 5, lane = threadIdx.x & 31;
  if (lane == 0) red[w] = ss;
  __syncthreads();
  if (w == 0) {
    float t = (lane < 8) ? red[lane] : 0.0f;
    #pragma unroll
    for (int o = 4; o > 0; o >>= 1) t += __shfl_xor(t, o, 32);
    if (lane == 0) invs = rsqrtf(t / 512.0f + 1e-6f);
  }
  __syncthreads();
  const float inv = invs;
  for (int i = threadIdx.x; i < 512; i += 256) {
    __bf16 b = f2bf(r[i] * inv * lnw[i]);
    kf[(long)s * 576 + i] = b;
    klT[(long)i * S + s] = b;
  }
  if (threadIdx.x < 64) {
    int i = threadIdx.x;
    float x = r[512 + i];
    float oth = (i < 32) ? -r[512 + i + 32] : r[512 + i - 32];
    float v = x * cosb[(long)s * 64 + i] + oth * sinb[(long)s * 64 + i];
    kf[(long)s * 576 + 512 + i] = f2bf(v);
  }
}

// ================================== q: split nope / rope pe into buffers
// q f32 (S,H,192) -> qnope bf16 (S,H,128), qf[...,512:576] bf16 roped
__global__ __launch_bounds__(256) void q_split_rope_kernel(
    const float* __restrict__ q, const float* __restrict__ cosb,
    const float* __restrict__ sinb, __bf16* __restrict__ qnope,
    __bf16* __restrict__ qf) {
  const int s = blockIdx.x;
  const float* r = q + (long)s * 3072;
  for (int e = threadIdx.x; e < 3072; e += 256) {
    int h = e / 192, c = e % 192;
    if (c < 128) {
      qnope[(long)s * 2048 + h * 128 + c] = f2bf(r[e]);
    } else {
      int i = c - 128;
      float x = r[e];
      float oth = (i < 32) ? -r[h * 192 + 128 + i + 32] : r[h * 192 + 128 + i - 32];
      float v = x * cosb[(long)s * 64 + i] + oth * sinb[(long)s * 64 + i];
      qf[(long)s * 9216 + h * 576 + 512 + i] = f2bf(v);
    }
  }
}

// ============================================================== GEMM
// C[M,N] = A[M,K](bf16,row-major,lda) * B, with B given transposed Bt[N][K].
// EPI 0: f32 out (+opt residual); 1: bf16 out; 2: dual-B silu(g)*u bf16 out.
template <int EPI>
__global__ __launch_bounds__(256) void gemm_bf16_kernel(
    const __bf16* __restrict__ A, long lda, long aH,
    const __bf16* __restrict__ Bt, long ldbt, long btH,
    const __bf16* __restrict__ Bt2,
    float* __restrict__ C, __bf16* __restrict__ Cb, long ldc, long cH,
    const float* __restrict__ res, int M, int N, int K) {
  constexpr int BM = 128;
  constexpr int BN = (EPI == 2) ? 64 : 128;
  constexpr int WN = (EPI == 2) ? 2 : 4;   // 16-wide tiles per wave (N dir)
  constexpr int LDA = 48;                  // padded LDS row (bf16 elems)
  __shared__ __bf16 sA[BM * LDA];
  __shared__ __bf16 sB[BN * LDA];
  __shared__ __bf16 sB2[(EPI == 2) ? BN * LDA : 16];

  const int tid = threadIdx.x, lane = tid & 31;
  const int w = tid >> 5, wm = w & 3, wn = w >> 2;
  const int z = blockIdx.z;
  A  += (long)z * aH;
  Bt += (long)z * btH;
  const __bf16* Bt2z = Bt2 + (long)z * btH;
  const long cOff = (long)z * cH;
  const int m0 = blockIdx.y * BM;
  const int n0 = blockIdx.x * BN;
  const int hi8 = (lane >> 4) << 3;

  v8f acc[2][WN];
  v8f acc2[(EPI == 2) ? 2 : 1][(EPI == 2) ? WN : 1];
  const v8f vz = {0.f, 0.f, 0.f, 0.f, 0.f, 0.f, 0.f, 0.f};
  #pragma unroll
  for (int i = 0; i < 2; ++i)
    #pragma unroll
    for (int j = 0; j < WN; ++j) acc[i][j] = vz;
  if (EPI == 2) {
    #pragma unroll
    for (int i = 0; i < 2; ++i)
      #pragma unroll
      for (int j = 0; j < WN; ++j) acc2[i][j] = vz;
  }

  const int kTiles = K >> 5;
  for (int kt = 0; kt < kTiles; ++kt) {
    const int k0 = kt << 5;
    __syncthreads();
    { // A tile 128x32: 2 threads/row, 16 bf16 each
      int r = tid >> 1, kc = (tid & 1) * 16, gm = m0 + r;
      uint4 v0 = {0, 0, 0, 0}, v1 = {0, 0, 0, 0};
      if (gm < M) {
        const uint4* g = reinterpret_cast<const uint4*>(A + (long)gm * lda + k0 + kc);
        v0 = g[0]; v1 = g[1];
        if (kt + 1 < kTiles) __builtin_prefetch(A + (long)gm * lda + k0 + 32 + kc, 0, 0);
      }
      uint4* sp = reinterpret_cast<uint4*>(&sA[r * LDA + kc]);
      sp[0] = v0; sp[1] = v1;
    }
    { // B tile BNx32
      constexpr int TPR = 256 / BN;
      constexpr int EPT = 32 / TPR;
      int r = tid / TPR, kc = (tid % TPR) * EPT, gn = n0 + r;
      const bool ok = (gn < N);
      uint4 v0 = {0, 0, 0, 0}, v1 = {0, 0, 0, 0};
      if (ok) {
        const uint4* g = reinterpret_cast<const uint4*>(Bt + (long)gn * ldbt + k0 + kc);
        v0 = g[0];
        if (EPT == 16) v1 = g[1];
      }
      uint4* sp = reinterpret_cast<uint4*>(&sB[r * LDA + kc]);
      sp[0] = v0;
      if (EPT == 16) sp[1] = v1;
      if (EPI == 2) {
        uint4 u0 = {0, 0, 0, 0};
        if (ok) u0 = *reinterpret_cast<const uint4*>(Bt2z + (long)gn * ldbt + k0 + kc);
        *reinterpret_cast<uint4*>(&sB2[r * LDA + kc]) = u0;
      }
    }
    __syncthreads();

    Frag aF[2];
    #pragma unroll
    for (int mi = 0; mi < 2; ++mi) {
      int row = wm * 32 + mi * 16 + (lane & 15);
      int kb = (lane >> 4) * 16;
      const uint4* sp = reinterpret_cast<const uint4*>(&sA[row * LDA + kb]);
      aF[mi].q[0] = sp[0]; aF[mi].q[1] = sp[1];
    }
    #pragma unroll
    for (int nj = 0; nj < WN; ++nj) {
      int nrow = wn * (BN / 2) + nj * 16 + (lane & 15);
      int kb2 = (lane >> 4) * 16;
      Frag bF;
      const uint4* sp = reinterpret_cast<const uint4*>(&sB[nrow * LDA + kb2]);
      bF.q[0] = sp[0]; bF.q[1] = sp[1];
      #pragma unroll
      for (int mi = 0; mi < 2; ++mi) acc[mi][nj] = wmma_bf16(aF[mi].v, bF.v, acc[mi][nj]);
      if (EPI == 2) {
        Frag cF;
        const uint4* sq = reinterpret_cast<const uint4*>(&sB2[nrow * LDA + kb2]);
        cF.q[0] = sq[0]; cF.q[1] = sq[1];
        #pragma unroll
        for (int mi = 0; mi < 2; ++mi) acc2[mi][nj] = wmma_bf16(aF[mi].v, cF.v, acc2[mi][nj]);
      }
    }
  }

  #pragma unroll
  for (int mi = 0; mi < 2; ++mi)
    #pragma unroll
    for (int nj = 0; nj < WN; ++nj)
      #pragma unroll
      for (int r = 0; r < 8; ++r) {
        int row = m0 + wm * 32 + mi * 16 + r + hi8;
        int col = n0 + wn * (BN / 2) + nj * 16 + (lane & 15);
        if (row < M && col < N) {
          long idx = (long)row * ldc + col + cOff;
          float v = acc[mi][nj][r];
          if (EPI == 0) {
            if (res) v += res[idx];
            C[idx] = v;
          } else if (EPI == 1) {
            Cb[idx] = f2bf(v);
          } else {
            float g = v, u = acc2[mi][nj][r];
            float sg = g / (1.0f + __expf(-g));
            Cb[idx] = f2bf(sg * u);
          }
        }
      }
}

// ========================================================= flash attention
// per block: one head, 32 query rows. Chunks of 512 keys, online softmax.
// qf (S,H,576) bf16 ; kf (S,576) bf16 ; klT (512,S) bf16 ; olat (S,H,512) bf16
#define ATT_CH   512
#define QT_LDA   584
#define P_LDA    520
#define SMEM_QT  0
#define SMEM_SC  37376
#define SMEM_PB  102912
#define SMEM_RM  136192
#define SMEM_RS  136320
#define SMEM_CF  136448
#define ATT_SMEM 136576

__global__ __launch_bounds__(256) void attn_kernel(
    const __bf16* __restrict__ qf, const __bf16* __restrict__ kf,
    const __bf16* __restrict__ klT, __bf16* __restrict__ olat, int S) {
  extern __shared__ char smem[];
  __bf16* qtile = (__bf16*)(smem + SMEM_QT);
  float*  sc    = (float*)(smem + SMEM_SC);
  __bf16* pb    = (__bf16*)(smem + SMEM_PB);
  float*  rmax  = (float*)(smem + SMEM_RM);
  float*  rsum  = (float*)(smem + SMEM_RS);
  float*  cfac  = (float*)(smem + SMEM_CF);

  const int tid = threadIdx.x, lane = tid & 31, w = tid >> 5;
  const int hi8 = (lane >> 4) << 3;
  const int h = blockIdx.y;
  const int s0 = blockIdx.x * 32;
  const int smax = s0 + 31;
  const float scale = 0.07216878364870322f; // (128+64)^-0.5

  // stage this head's 32x576 q rows into LDS
  for (int i = tid; i < 32 * 72; i += 256) {
    int row = i / 72, c8 = (i % 72) * 8;
    const uint4* g =
        reinterpret_cast<const uint4*>(qf + ((long)(s0 + row) * 16 + h) * 576 + c8);
    *reinterpret_cast<uint4*>(&qtile[row * QT_LDA + c8]) = *g;
  }
  if (tid < 32) { rmax[tid] = -3e38f; rsum[tid] = 0.0f; }
  __syncthreads();

  v8f acc[2][4];
  const v8f vz = {0.f, 0.f, 0.f, 0.f, 0.f, 0.f, 0.f, 0.f};
  #pragma unroll
  for (int i = 0; i < 2; ++i)
    #pragma unroll
    for (int j = 0; j < 4; ++j) acc[i][j] = vz;

  const int nChunks = (smax + ATT_CH) / ATT_CH;
  for (int ch = 0; ch < nChunks; ++ch) {
    const int c0 = ch * ATT_CH;
    // ---- scores: sc[m][t-c0] = scale * qf . kf  (causal-masked) ----
    for (int T = w; T < 2 * (ATT_CH / 16); T += 8) {
      int mi = T & 1, j = T >> 1;
      int t0 = c0 + j * 16;
      if (t0 > smax) {
        #pragma unroll
        for (int r = 0; r < 8; ++r)
          sc[(mi * 16 + r + hi8) * ATT_CH + j * 16 + (lane & 15)] = -1e30f;
      } else {
        v8f a = vz;
        const int arow = mi * 16 + (lane & 15);
        const int kb = (lane >> 4) * 16;
        const int trow = t0 + (lane & 15);
        const int kb2 = (lane >> 4) * 16;
        #pragma unroll
        for (int kc = 0; kc < 18; ++kc) {
          Frag aF;
          const uint4* sa =
              reinterpret_cast<const uint4*>(&qtile[arow * QT_LDA + kc * 32 + kb]);
          aF.q[0] = sa[0]; aF.q[1] = sa[1];
          Frag bF;
          const uint4* gb =
              reinterpret_cast<const uint4*>(kf + (long)trow * 576 + kc * 32 + kb2);
          bF.q[0] = gb[0]; bF.q[1] = gb[1];
          a = wmma_bf16(aF.v, bF.v, a);
        }
        #pragma unroll
        for (int r = 0; r < 8; ++r) {
          int m = mi * 16 + r + hi8;
          int t = t0 + (lane & 15);
          sc[m * ATT_CH + (t - c0)] = (t <= s0 + m) ? a[r] * scale : -1e30f;
        }
      }
    }
    __syncthreads();
    // ---- online softmax per row ----
    for (int m = w; m < 32; m += 8) {
      float mx = -3e38f;
      for (int i = lane; i < ATT_CH; i += 32) mx = fmaxf(mx, sc[m * ATT_CH + i]);
      #pragma unroll
      for (int o = 16; o > 0; o >>= 1) mx = fmaxf(mx, __shfl_xor(mx, o, 32));
      float om = rmax[m];
      float nm = fmaxf(om, mx);
      float sum = 0.0f;
      for (int i = lane; i < ATT_CH; i += 32) {
        float e = __expf(sc[m * ATT_CH + i] - nm);
        pb[m * P_LDA + i] = f2bf(e);
        sum += e;
      }
      #pragma unroll
      for (int o = 16; o > 0; o >>= 1) sum += __shfl_xor(sum, o, 32);
      if (lane == 0) {
        float cf = __expf(om - nm);
        cfac[m] = cf;
        rsum[m] = rsum[m] * cf + sum;
        rmax[m] = nm;
      }
    }
    __syncthreads();
    // ---- rescale running acc, then acc += P * k_lat_chunk ----
    #pragma unroll
    for (int mi = 0; mi < 2; ++mi) {
      float cv[8];
      #pragma unroll
      for (int r = 0; r < 8; ++r) cv[r] = cfac[mi * 16 + r + hi8];
      #pragma unroll
      for (int nj = 0; nj < 4; ++nj)
        #pragma unroll
        for (int r = 0; r < 8; ++r) acc[mi][nj][r] *= cv[r];
    }
    int lim = smax + 1 - c0; if (lim > ATT_CH) lim = ATT_CH;
    const int kkMax = (lim + 31) >> 5;
    for (int kk = 0; kk < kkMax; ++kk) {
      Frag aF[2];
      const int kb = (lane >> 4) * 16;
      #pragma unroll
      for (int mi = 0; mi < 2; ++mi) {
        const uint4* sp = reinterpret_cast<const uint4*>(
            &pb[(mi * 16 + (lane & 15)) * P_LDA + kk * 32 + kb]);
        aF[mi].q[0] = sp[0]; aF[mi].q[1] = sp[1];
      }
      const int kb2 = (lane >> 4) * 16;
      #pragma unroll
      for (int nj = 0; nj < 4; ++nj) {
        int n = w * 64 + nj * 16 + (lane & 15);
        Frag bF;
        const uint4* gb =
            reinterpret_cast<const uint4*>(klT + (long)n * S + c0 + kk * 32 + kb2);
        bF.q[0] = gb[0]; bF.q[1] = gb[1];
        #pragma unroll
        for (int mi = 0; mi < 2; ++mi) acc[mi][nj] = wmma_bf16(aF[mi].v, bF.v, acc[mi][nj]);
      }
    }
    __syncthreads();
  }
  // ---- normalize by running sum, write o_lat bf16 ----
  #pragma unroll
  for (int mi = 0; mi < 2; ++mi)
    #pragma unroll
    for (int r = 0; r < 8; ++r) {
      int m = mi * 16 + r + hi8;
      float inv = 1.0f / rsum[m];
      int s = s0 + m;
      #pragma unroll
      for (int nj = 0; nj < 4; ++nj) {
        int n = w * 64 + nj * 16 + (lane & 15);
        olat[((long)s * 16 + h) * 512 + n] = f2bf(acc[mi][nj][r] * inv);
      }
    }
}

// ================================================================= launch
extern "C" void kernel_launch(void* const* d_in, const int* in_sizes, int n_in,
                              void* d_out, int out_size, void* d_ws, size_t ws_size,
                              hipStream_t stream) {
  (void)in_sizes; (void)n_in; (void)out_size; (void)ws_size;
  const float* x_in   = (const float*)d_in[0];
  const float* cosb   = (const float*)d_in[1];
  const float* sinb   = (const float*)d_in[2];
  const float* ln1_w  = (const float*)d_in[3];
  const float* q_a_w  = (const float*)d_in[4];
  const float* q_a_ln = (const float*)d_in[5];
  const float* q_b_w  = (const float*)d_in[6];
  const float* kv_a_w = (const float*)d_in[7];
  const float* kv_ln  = (const float*)d_in[8];
  const float* kc_w   = (const float*)d_in[9];
  const float* vc_w   = (const float*)d_in[10];
  const float* o_w    = (const float*)d_in[11];
  const float* ln2_w  = (const float*)d_in[12];
  const float* gu_w   = (const float*)d_in[13];
  const float* dn_w   = (const float*)d_in[14];
  float* out = (float*)d_out;

  char* ws = (char*)d_ws;
  size_t off = 0;
  auto alloc = [&](size_t bytes) -> char* {
    char* p = ws + off;
    off += (bytes + 255) & ~(size_t)255;
    return p;
  };
  // weights (bf16, transposed to [N][K])
  __bf16* W_qaT  = (__bf16*)alloc(1536L * 2048 * 2);
  __bf16* W_qbT  = (__bf16*)alloc(3072L * 1536 * 2);
  __bf16* W_kvaT = (__bf16*)alloc(576L * 2048 * 2);
  __bf16* W_kcT  = (__bf16*)alloc(16L * 512 * 128 * 2);
  __bf16* W_vcT  = (__bf16*)alloc(16L * 128 * 512 * 2);
  __bf16* W_oT   = (__bf16*)alloc(2048L * 2048 * 2);
  __bf16* W_guT  = (__bf16*)alloc(16384L * 2048 * 2);
  __bf16* W_dnT  = (__bf16*)alloc(2048L * 8192 * 2);
  // activations
  __bf16* HBF   = (__bf16*)alloc(2048L * 2048 * 2);
  float*  QAF   = (float*) alloc(2048L * 1536 * 4);
  __bf16* QAN   = (__bf16*)alloc(2048L * 1536 * 2);
  float*  QFULL = (float*) alloc(2048L * 3072 * 4);
  __bf16* QNOPE = (__bf16*)alloc(2048L * 2048 * 2);
  __bf16* QF    = (__bf16*)alloc(2048L * 16 * 576 * 2);
  float*  KV    = (float*) alloc(2048L * 576 * 4);
  __bf16* KF    = (__bf16*)alloc(2048L * 576 * 2);
  __bf16* KLT   = (__bf16*)alloc(512L * 2048 * 2);
  __bf16* OLAT  = (__bf16*)alloc(2048L * 16 * 512 * 2);
  __bf16* OB    = (__bf16*)alloc(2048L * 2048 * 2);
  float*  HID   = (float*) alloc(2048L * 2048 * 4);
  __bf16* X2    = (__bf16*)alloc(2048L * 2048 * 2);
  __bf16* MLPH  = (__bf16*)alloc(2048L * 8192 * 2);

  const dim3 blk(256);
  auto tg = [](int R, int C, int B) { return dim3((C + 31) / 32, (R + 31) / 32, B); };

  // --- weight convert+transpose ---
  transpose_f32_bf16<<<tg(2048, 1536, 1), blk, 0, stream>>>(q_a_w, W_qaT, 2048, 1536);
  transpose_f32_bf16<<<tg(1536, 3072, 1), blk, 0, stream>>>(q_b_w, W_qbT, 1536, 3072);
  transpose_f32_bf16<<<tg(2048, 576, 1),  blk, 0, stream>>>(kv_a_w, W_kvaT, 2048, 576);
  transpose_f32_bf16<<<tg(128, 512, 16),  blk, 0, stream>>>(kc_w, W_kcT, 128, 512);
  transpose_f32_bf16<<<tg(512, 128, 16),  blk, 0, stream>>>(vc_w, W_vcT, 512, 128);
  transpose_f32_bf16<<<tg(2048, 2048, 1), blk, 0, stream>>>(o_w, W_oT, 2048, 2048);
  transpose_f32_bf16<<<tg(2048, 16384, 1),blk, 0, stream>>>(gu_w, W_guT, 2048, 16384);
  transpose_f32_bf16<<<tg(8192, 2048, 1), blk, 0, stream>>>(dn_w, W_dnT, 8192, 2048);

  // --- attention path ---
  rmsnorm_bf16_kernel<<<2048, blk, 0, stream>>>(x_in, ln1_w, HBF, 2048);
  gemm_bf16_kernel<0><<<dim3(12, 16, 1), blk, 0, stream>>>(
      HBF, 2048, 0, W_qaT, 2048, 0, nullptr, QAF, nullptr, 1536, 0, nullptr,
      2048, 1536, 2048);
  rmsnorm_bf16_kernel<<<2048, blk, 0, stream>>>(QAF, q_a_ln, QAN, 1536);
  gemm_bf16_kernel<0><<<dim3(24, 16, 1), blk, 0, stream>>>(
      QAN, 1536, 0, W_qbT, 1536, 0, nullptr, QFULL, nullptr, 3072, 0, nullptr,
      2048, 3072, 1536);
  gemm_bf16_kernel<0><<<dim3(5, 16, 1), blk, 0, stream>>>(
      HBF, 2048, 0, W_kvaT, 2048, 0, nullptr, KV, nullptr, 576, 0, nullptr,
      2048, 576, 2048);
  q_split_rope_kernel<<<2048, blk, 0, stream>>>(QFULL, cosb, sinb, QNOPE, QF);
  kv_norm_rope_kernel<<<2048, blk, 0, stream>>>(KV, kv_ln, cosb, sinb, KF, KLT, 2048);
  // q_lat per head -> qf[...,0:512]
  gemm_bf16_kernel<1><<<dim3(4, 16, 16), blk, 0, stream>>>(
      QNOPE, 2048, 128, W_kcT, 128, 512L * 128, nullptr, nullptr, QF, 9216, 576,
      nullptr, 2048, 512, 128);
  // flash attention
  (void)hipFuncSetAttribute((const void*)attn_kernel,
                            hipFuncAttributeMaxDynamicSharedMemorySize, ATT_SMEM);
  attn_kernel<<<dim3(64, 16, 1), blk, ATT_SMEM, stream>>>(QF, KF, KLT, OLAT, 2048);
  // o = o_lat @ vc  per head
  gemm_bf16_kernel<1><<<dim3(1, 16, 16), blk, 0, stream>>>(
      OLAT, 8192, 512, W_vcT, 512, 128L * 512, nullptr, nullptr, OB, 2048, 128,
      nullptr, 2048, 128, 512);
  // hidden = residual + o @ o_w
  gemm_bf16_kernel<0><<<dim3(16, 16, 1), blk, 0, stream>>>(
      OB, 2048, 0, W_oT, 2048, 0, nullptr, HID, nullptr, 2048, 0, x_in,
      2048, 2048, 2048);

  // --- MLP path ---
  rmsnorm_bf16_kernel<<<2048, blk, 0, stream>>>(HID, ln2_w, X2, 2048);
  gemm_bf16_kernel<2><<<dim3(128, 16, 1), blk, 0, stream>>>(
      X2, 2048, 0, W_guT, 2048, 0, W_guT + 8192L * 2048, nullptr, MLPH, 8192, 0,
      nullptr, 2048, 8192, 2048);
  gemm_bf16_kernel<0><<<dim3(16, 16, 1), blk, 0, stream>>>(
      MLPH, 8192, 0, W_dnT, 8192, 0, nullptr, out, nullptr, 2048, 0, HID,
      2048, 2048, 8192);
}
